// ModelWithEdgeFeatures_73375221285170
// MI455X (gfx1250) — matrix operand order repaired and checked by reading, not compiled, not run on hardware
//
#include <hip/hip_runtime.h>
#include <stdint.h>

typedef __attribute__((ext_vector_type(16))) _Float16 v16h;
typedef __attribute__((ext_vector_type(8)))  _Float16 v8h;
typedef __attribute__((ext_vector_type(8)))  float    v8f;

#define NNODES 20000
#define NEDGES 320000
#define NPAD   20096   // ceil(20000/128)*128
#define OUTF   256
#define EPSV   1e-5f

// Async global->LDS path (CDNA5 ASYNCcnt), guarded so absence falls back cleanly.
#if defined(__has_builtin)
#  if __has_builtin(__builtin_amdgcn_global_load_async_to_lds_b128) && \
      __has_builtin(__builtin_amdgcn_s_wait_asynccnt)
#    define USE_ASYNC_LDS 1
#  endif
#endif
#ifndef USE_ASYNC_LDS
#  define USE_ASYNC_LDS 0
#endif

#if USE_ASYNC_LDS
typedef int v4i __attribute__((vector_size(16)));
typedef __attribute__((address_space(1))) v4i* GPTR;   // global (__device__) int4*
typedef __attribute__((address_space(3))) v4i* LPTR;   // LDS (__shared__) int4*
#endif

// ---------------- utility kernels ----------------

// W (fp32, [256][K]) -> Wh (f16, [256][Kpad], zero-padded in K)
__global__ void k_convert_w(const float* __restrict__ W, _Float16* __restrict__ Wh,
                            int K, int Kpad) {
  int t = blockIdx.x * blockDim.x + threadIdx.x;
  int total = OUTF * Kpad;
  if (t >= total) return;
  int n = t / Kpad, k = t - n * Kpad;
  float v = (k < K) ? W[n * K + k] : 0.f;
  Wh[t] = (_Float16)v;
}

// self-loop baseline: deg=1, ea_agg=1
__global__ void k_init_deg_ea(float* __restrict__ deg, float* __restrict__ ea_agg) {
  int t = blockIdx.x * blockDim.x + threadIdx.x;
  if (t < NNODES) deg[t] = 1.0f;
  if (t < NNODES * 16) ea_agg[t] = 1.0f;
}

__global__ void k_deg_scatter(const int* __restrict__ ei, float* __restrict__ deg) {
  int e = blockIdx.x * blockDim.x + threadIdx.x;
  if (e >= NEDGES) return;
  int d = ei[NEDGES + e];
  atomicAdd(&deg[d], 1.0f);
}

__global__ void k_ea_scatter(const int* __restrict__ ei, const float* __restrict__ ea,
                             float* __restrict__ ea_agg) {
  int t = blockIdx.x * blockDim.x + threadIdx.x;
  if (t >= NEDGES * 16) return;
  int e = t >> 4, k = t & 15;
  int d = ei[NEDGES + e];
  atomicAdd(&ea_agg[d * 16 + k], ea[t]);
}

// agg[i][:] = h[i][:]  (self-loop contribution), float4 copies
__global__ void k_init_agg(const float* __restrict__ h, float* __restrict__ agg, int total4) {
  int t = blockIdx.x * blockDim.x + threadIdx.x;
  if (t >= total4) return;
  ((float4*)agg)[t] = ((const float4*)h)[t];
}

// agg[dst][:] += h[src][:]  (one thread per edge per 4 features)
__global__ void k_h_scatter(const int* __restrict__ ei, const float* __restrict__ h,
                            float* __restrict__ agg, int Fq /*F/4*/) {
  int t = blockIdx.x * blockDim.x + threadIdx.x;
  if (t >= NEDGES * Fq) return;
  int e = t / Fq, q = t - e * Fq;
  int s = ei[e];
  int d = ei[NEDGES + e];
  float4 v = ((const float4*)h)[s * Fq + q];
  float* a = agg + (size_t)d * (size_t)(Fq * 4) + q * 4;
  atomicAdd(a + 0, v.x); atomicAdd(a + 1, v.y);
  atomicAdd(a + 2, v.z); atomicAdd(a + 3, v.w);
}

// Xin[i][:] = f16([deg_i*h_i | agg_i | ea_agg_i | 0-pad]) ; pad rows >= N are zero
__global__ void k_pack(const float* __restrict__ h, const float* __restrict__ agg,
                       const float* __restrict__ ea_agg, const float* __restrict__ deg,
                       _Float16* __restrict__ Xin, int F, int Kpad) {
  int t = blockIdx.x * blockDim.x + threadIdx.x;
  int total = NPAD * Kpad;
  if (t >= total) return;
  int i = t / Kpad, j = t - i * Kpad;
  float v = 0.f;
  if (i < NNODES) {
    if (j < F)               v = h[i * F + j] * deg[i];
    else if (j < 2 * F)      v = agg[i * F + (j - F)];
    else if (j < 2 * F + 16) v = ea_agg[i * 16 + (j - 2 * F)];
  }
  Xin[t] = (_Float16)v;
}

// ---------------- WMMA GEMM + fused epilogue ----------------
// out[N][256] = relu( Xin @ Wh^T + deg*b )  [+ BN + relu when do_bn]
// Block: 256 thr = 8 waves; tile 128(M) x 128(N); K-step 32.
// Async path: double-buffered LDS, GLOBAL_LOAD_ASYNC_TO_LDS_B128 prefetch of
// tile k+1 overlapped with WMMA of tile k. Sync fallback: register prefetch.
__launch_bounds__(256)
__global__ void k_gemm_wmma(const _Float16* __restrict__ Xin, const _Float16* __restrict__ Wh,
                            const float* __restrict__ deg, const float* __restrict__ bias,
                            const float* __restrict__ g, const float* __restrict__ beta,
                            const float* __restrict__ rm, const float* __restrict__ rv,
                            float* __restrict__ out, int Kpad, int do_bn) {
#if USE_ASYNC_LDS
  __shared__ _Float16 lsA[2][128 * 32];
  __shared__ _Float16 lsB[2][128 * 32];
#else
  __shared__ _Float16 lsA[1][128 * 32];
  __shared__ _Float16 lsB[1][128 * 32];
#endif

  const int tid   = threadIdx.x;
  const int lane  = tid & 31;
  const int wave  = tid >> 5;     // 0..7  (wave32)
  const int waveM = wave & 3;     // 4 sub-tiles of 32 rows
  const int waveN = wave >> 2;    // 2 sub-tiles of 64 cols
  const int mBase = blockIdx.x * 128;
  const int nBase = blockIdx.y * 128;

  v8f acc[2][4];
  #pragma unroll
  for (int t = 0; t < 2; ++t)
    #pragma unroll
    for (int c = 0; c < 4; ++c) { v8f z = {}; acc[t][c] = z; }

  const int ldRow = tid >> 1;         // 0..127
  const int ldSeg = (tid & 1) * 16;   // halves within the 32-wide K slab

  const int half  = lane >> 4;        // 0 or 1
  const int mn    = lane & 15;
  const int aoff0 = half * 8;         // A frag: K {0..7}/{8..15}
  const int aoff1 = 16 + half * 8;    // A frag: K {16..23}/{24..31}
  const int boff  = half * 8;         // B frag: K {0..7,16..23}/{8..15,24..31}

  // fragment gather + 8 WMMAs for one staged 32-wide K slab
  auto compute = [&](const _Float16* A, const _Float16* B) {
    v16h afrag[2];
    #pragma unroll
    for (int t = 0; t < 2; ++t) {
      const _Float16* base = A + (waveM * 32 + t * 16 + mn) * 32;
      v8h lo = *(const v8h*)(base + aoff0);
      v8h hi = *(const v8h*)(base + aoff1);
      #pragma unroll
      for (int i = 0; i < 8; ++i) { afrag[t][i] = lo[i]; afrag[t][i + 8] = hi[i]; }
    }
    v16h bfrag[4];
    #pragma unroll
    for (int c = 0; c < 4; ++c) {
      const _Float16* base = B + (waveN * 64 + c * 16 + mn) * 32;
      v8h lo = *(const v8h*)(base + boff);
      v8h hi = *(const v8h*)(base + boff + 16);
      #pragma unroll
      for (int i = 0; i < 8; ++i) { bfrag[c][i] = lo[i]; bfrag[c][i + 8] = hi[i]; }
    }
    #pragma unroll
    for (int t = 0; t < 2; ++t)
      #pragma unroll
      for (int c = 0; c < 4; ++c)
        acc[t][c] = __builtin_amdgcn_wmma_f32_16x16x32_f16(
            false, afrag[t], false, bfrag[c], (short)0, acc[t][c], false, false);
  };

#if USE_ASYNC_LDS
  // ---- double-buffered async pipeline ----
  auto prefetch = [&](int buf, int k0) {
    const _Float16* ga = Xin + (size_t)(mBase + ldRow) * Kpad + k0 + ldSeg;
    const _Float16* gb = Wh  + (size_t)(nBase + ldRow) * Kpad + k0 + ldSeg;
    _Float16* la = &lsA[buf][ldRow * 32 + ldSeg];
    _Float16* lb = &lsB[buf][ldRow * 32 + ldSeg];
    __builtin_amdgcn_global_load_async_to_lds_b128((GPTR)ga,       (LPTR)la,       0, 0);
    __builtin_amdgcn_global_load_async_to_lds_b128((GPTR)(ga + 8), (LPTR)(la + 8), 0, 0);
    __builtin_amdgcn_global_load_async_to_lds_b128((GPTR)gb,       (LPTR)lb,       0, 0);
    __builtin_amdgcn_global_load_async_to_lds_b128((GPTR)(gb + 8), (LPTR)(lb + 8), 0, 0);
  };

  int buf = 0;
  prefetch(0, 0);
  for (int k0 = 0; k0 < Kpad; k0 += 32) {
    const bool hasNext = (k0 + 32) < Kpad;
    __syncthreads();                      // readers of buf^1 (iter k-1) are done
    if (hasNext) {
      prefetch(buf ^ 1, k0 + 32);         // overlap next tile with this tile's math
      __builtin_amdgcn_s_wait_asynccnt(4); // in-order: current tile's 4 loads landed
    } else {
      __builtin_amdgcn_s_wait_asynccnt(0);
    }
    __syncthreads();                      // every wave's current-tile data visible
    compute(lsA[buf], lsB[buf]);
    buf ^= 1;
  }
#else
  // ---- sync fallback: register prefetch of tile k+1 before tile k's math ----
  const _Float16* ga = Xin + (size_t)(mBase + ldRow) * Kpad + ldSeg;
  const _Float16* gb = Wh  + (size_t)(nBase + ldRow) * Kpad + ldSeg;
  v8h a0 = *(const v8h*)(ga);     v8h a1 = *(const v8h*)(ga + 8);
  v8h b0 = *(const v8h*)(gb);     v8h b1 = *(const v8h*)(gb + 8);
  for (int k0 = 0; k0 < Kpad; k0 += 32) {
    __syncthreads();                      // prior iteration's LDS reads complete
    *(v8h*)(&lsA[0][ldRow * 32 + ldSeg])     = a0;
    *(v8h*)(&lsA[0][ldRow * 32 + ldSeg + 8]) = a1;
    *(v8h*)(&lsB[0][ldRow * 32 + ldSeg])     = b0;
    *(v8h*)(&lsB[0][ldRow * 32 + ldSeg + 8]) = b1;
    __syncthreads();
    if (k0 + 32 < Kpad) {                 // issue next tile's loads, wait deferred
      const _Float16* na = ga + (k0 + 32);
      const _Float16* nb = gb + (k0 + 32);
      a0 = *(const v8h*)(na); a1 = *(const v8h*)(na + 8);
      b0 = *(const v8h*)(nb); b1 = *(const v8h*)(nb + 8);
    }
    compute(lsA[0], lsB[0]);
  }
#endif

  // epilogue: D layout -> row = r + 8*(lane>=16), col = lane&15
  const int rowoff = (lane >> 4) * 8;
  const int col0   = lane & 15;
  #pragma unroll
  for (int t = 0; t < 2; ++t) {
    #pragma unroll
    for (int c = 0; c < 4; ++c) {
      const int col  = nBase + waveN * 64 + c * 16 + col0;
      const float bc = bias[col];
      float gs = 0.f, bt = 0.f, mean = 0.f, inv = 1.f;
      if (do_bn) { gs = g[col]; bt = beta[col]; mean = rm[col]; inv = rsqrtf(rv[col] + EPSV); }
      #pragma unroll
      for (int r = 0; r < 8; ++r) {
        int row = mBase + waveM * 32 + t * 16 + rowoff + r;
        if (row < NNODES) {
          float v = acc[t][c][r] + deg[row] * bc;
          v = fmaxf(v, 0.f);
          if (do_bn) { v = (v - mean) * inv * gs + bt; v = fmaxf(v, 0.f); }
          out[(size_t)row * OUTF + col] = v;
        }
      }
    }
  }
}

// ---------------- host launcher ----------------
extern "C" void kernel_launch(void* const* d_in, const int* in_sizes, int n_in,
                              void* d_out, int out_size, void* d_ws, size_t ws_size,
                              hipStream_t stream) {
  (void)in_sizes; (void)n_in; (void)out_size; (void)ws_size;
  const float* x   = (const float*)d_in[0];
  const int*   ei  = (const int*)d_in[1];
  const float* ea  = (const float*)d_in[2];
  const float* W[3]  = { (const float*)d_in[3], (const float*)d_in[5], (const float*)d_in[7] };
  const float* bi[3] = { (const float*)d_in[4], (const float*)d_in[6], (const float*)d_in[8] };
  const float* g0  = (const float*)d_in[9],  *be0 = (const float*)d_in[10];
  const float* rm0 = (const float*)d_in[11], *rv0 = (const float*)d_in[12];
  const float* g1  = (const float*)d_in[13], *be1 = (const float*)d_in[14];
  const float* rm1 = (const float*)d_in[15], *rv1 = (const float*)d_in[16];

  char* ws = (char*)d_ws;
  size_t off = 0;
  auto alloc = [&](size_t bytes) -> void* {
    void* p = ws + off;
    off = (off + bytes + 255) & ~(size_t)255;
    return p;
  };
  float*    deg    = (float*)alloc((size_t)NNODES * 4);
  float*    ea_agg = (float*)alloc((size_t)NNODES * 16 * 4);
  float*    agg    = (float*)alloc((size_t)NNODES * 256 * 4);
  _Float16* Xin    = (_Float16*)alloc((size_t)NPAD * 544 * 2);
  _Float16* Wh     = (_Float16*)alloc((size_t)OUTF * 544 * 2);
  float*    h0     = (float*)alloc((size_t)NNODES * 256 * 4);
  float*    h1     = (float*)alloc((size_t)NNODES * 256 * 4);

  const int B = 256;
  auto cdiv = [](int a, int b) { return (a + b - 1) / b; };

  // once: degrees and edge-attr aggregation (reused by all 3 layers)
  k_init_deg_ea<<<cdiv(NNODES * 16, B), B, 0, stream>>>(deg, ea_agg);
  k_deg_scatter<<<cdiv(NEDGES, B), B, 0, stream>>>(ei, deg);
  k_ea_scatter<<<cdiv(NEDGES * 16, B), B, 0, stream>>>(ei, ea, ea_agg);

  struct LayerCfg { const float* h; int F; float* out; int do_bn;
                    const float *g, *be, *rm, *rv; };
  LayerCfg L[3] = {
    { x,  128, h0,            1, g0, be0, rm0, rv0 },
    { h0, 256, h1,            1, g1, be1, rm1, rv1 },
    { h1, 256, (float*)d_out, 0, g0, be0, rm0, rv0 },  // bn ptrs unused when do_bn=0
  };

  for (int l = 0; l < 3; ++l) {
    const int F    = L[l].F;
    const int K    = 2 * F + 16;
    const int Kpad = ((K + 31) / 32) * 32;   // 288 or 544
    const int Fq   = F / 4;

    k_convert_w<<<cdiv(OUTF * Kpad, B), B, 0, stream>>>(W[l], Wh, K, Kpad);
    k_init_agg<<<cdiv(NNODES * Fq, B), B, 0, stream>>>(L[l].h, agg, NNODES * Fq);
    k_h_scatter<<<cdiv(NEDGES * Fq, B), B, 0, stream>>>(ei, L[l].h, agg, Fq);
    k_pack<<<cdiv(NPAD * Kpad, B), B, 0, stream>>>(L[l].h, agg, ea_agg, deg, Xin, F, Kpad);
    k_gemm_wmma<<<dim3(NPAD / 128, OUTF / 128), 256, 0, stream>>>(
        Xin, Wh, deg, bi[l], L[l].g, L[l].be, L[l].rm, L[l].rv, L[l].out, Kpad, L[l].do_bn);
  }
}
